// MoELayer_15187004358937
// MI455X (gfx1250) — matrix operand order repaired and checked by reading, not compiled, not run on hardware
//
#include <hip/hip_runtime.h>
#include <cstdint>
#include <cstddef>

// ---------------- problem dims ----------------
constexpr int E_  = 16;
constexpr int H_  = 2560;
constexpr int I_  = 1664;
constexpr int IS_ = 3328;
constexpr int T_  = 2048;   // B*S

// ---------------- CDNA5 WMMA types ----------------
typedef __attribute__((ext_vector_type(16))) __bf16   bf16x16;
typedef __attribute__((ext_vector_type(2)))  __bf16   bf16x2;
typedef __attribute__((ext_vector_type(8)))  float    f32x8;
typedef __attribute__((ext_vector_type(4)))  float    f32x4;
typedef __attribute__((ext_vector_type(4)))  unsigned u32x4;
typedef __attribute__((ext_vector_type(2)))  unsigned u32x2;

union Frag16 { u32x4 u[2]; bf16x16 v; };

// single f32 -> bf16 (RNE) via native conversion; backend picks v_cvt_*bf16_f32
__device__ __forceinline__ unsigned short f2bf(float f) {
  __bf16 b = (__bf16)f;
  return __builtin_bit_cast(unsigned short, b);
}

// packed f32x2 -> bf16x2 (one dword); prefer the packed builtin when present,
// otherwise two native truncations (backend fuses to v_cvt_pk_bf16_f32).
__device__ __forceinline__ unsigned pack2bf(float lo, float hi) {
#if __has_builtin(__builtin_amdgcn_cvt_pk_bf16_f32)
  bf16x2 p = __builtin_amdgcn_cvt_pk_bf16_f32(lo, hi);
  return __builtin_bit_cast(unsigned, p);
#else
  bf16x2 p;
  p[0] = (__bf16)lo;
  p[1] = (__bf16)hi;
  return __builtin_bit_cast(unsigned, p);
#endif
}

__device__ __forceinline__ float silu_f(float g) { return g / (1.0f + __expf(-g)); }

// ---------------- router: logits -> softmax -> top2 -> combine matrix ----------------
__global__ void router_kernel(const float* __restrict__ x,
                              const float* __restrict__ rw,
                              float* __restrict__ wcomb) {
  const int lane = threadIdx.x & 31;
  const int wave = threadIdx.x >> 5;
  const int tok  = blockIdx.x * (blockDim.x >> 5) + wave;
  if (tok >= T_) return;

  float acc[E_];
#pragma unroll
  for (int e = 0; e < E_; ++e) acc[e] = 0.0f;
  for (int h = lane; h < H_; h += 32) {
    const float xv = x[(size_t)tok * H_ + h];
#pragma unroll
    for (int e = 0; e < E_; ++e) acc[e] += xv * rw[(size_t)e * H_ + h];
  }
#pragma unroll
  for (int e = 0; e < E_; ++e)
#pragma unroll
    for (int off = 16; off > 0; off >>= 1)
      acc[e] += __shfl_xor(acc[e], off, 32);

  float mx = acc[0];
#pragma unroll
  for (int e = 1; e < E_; ++e) mx = fmaxf(mx, acc[e]);
  float p[E_]; float sum = 0.0f;
#pragma unroll
  for (int e = 0; e < E_; ++e) { p[e] = __expf(acc[e] - mx); sum += p[e]; }
  const float inv = 1.0f / sum;
#pragma unroll
  for (int e = 0; e < E_; ++e) p[e] *= inv;

  // top-2 (lowest index wins ties, matching lax.top_k)
  int i1 = 0; float v1 = p[0];
#pragma unroll
  for (int e = 1; e < E_; ++e) if (p[e] > v1) { v1 = p[e]; i1 = e; }
  int i2 = (i1 == 0) ? 1 : 0; float v2 = p[i2];
#pragma unroll
  for (int e = 0; e < E_; ++e) if (e != i1 && e != i2 && p[e] > v2) { v2 = p[e]; i2 = e; }
  const float s = v1 + v2;

  if (lane == 0) {
#pragma unroll
    for (int e = 0; e < E_; ++e) wcomb[(size_t)tok * E_ + e] = 0.0f;
    wcomb[(size_t)tok * E_ + i1] = v1 / s;
    wcomb[(size_t)tok * E_ + i2] = v2 / s;
  }
}

// ------------- deterministic per-expert token compaction (1 wave / expert) -------------
__global__ void build_lists_kernel(const float* __restrict__ wcomb,
                                   int* __restrict__ counts,
                                   int* __restrict__ toks,
                                   float* __restrict__ tws) {
  const int e = blockIdx.x;
  const int lane = threadIdx.x;
  int cnt = 0;
  for (int base = 0; base < T_; base += 32) {
    const int t = base + lane;
    const float w = wcomb[(size_t)t * E_ + e];
    const bool sel = (w > 0.0f);
    const unsigned m = (unsigned)__ballot(sel);
    const int pos = cnt + __popc(m & ((1u << lane) - 1u));
    if (sel) { toks[(size_t)e * T_ + pos] = t; tws[(size_t)e * T_ + pos] = w; }
    cnt += __popc(m);
  }
  const int padded = (cnt + 15) & ~15;            // pad tile to 16 with weight-0 tokens
  for (int i = cnt + lane; i < padded; i += 32) {
    toks[(size_t)e * T_ + i] = 0; tws[(size_t)e * T_ + i] = 0.0f;
  }
  if (lane == 0) counts[e] = cnt;
}

// ---- weight tile staging: 32(K) x 128(N) fp32 -> bf16, transposed, K-pairs packed ----
// thread slot s (0..511): r2 = s>>5 selects K row-pair (2*r2, 2*r2+1), c4 = s&31 selects
// 4 N-columns. Loads are float4 row-segments (fully coalesced); stores are packed
// bf16x2 dwords at Wt[c*32 + 2*r2] -- exactly the WMMA B-fragment K-pair layout.
template<int LD>
__device__ __forceinline__ void load_pair(const float* __restrict__ W, int k, int nb,
                                          int tid, f32x4 (&p)[2][2]) {
#pragma unroll
  for (int it = 0; it < 2; ++it) {
    const int s = tid + it * 256;
    const int r2 = s >> 5, c4 = s & 31;
    const size_t base = (size_t)(k + 2 * r2) * LD + nb + c4 * 4;
    p[it][0] = *(const f32x4*)(W + base);
    p[it][1] = *(const f32x4*)(W + base + LD);
  }
}

__device__ __forceinline__ void store_tr(unsigned short* __restrict__ Wt, int tid,
                                         const f32x4 (&p)[2][2]) {
#pragma unroll
  for (int it = 0; it < 2; ++it) {
    const int s = tid + it * 256;
    const int r2 = s >> 5, c4 = s & 31;
#pragma unroll
    for (int j = 0; j < 4; ++j) {
      const unsigned d = pack2bf(p[it][0][j], p[it][1][j]);
      *(unsigned*)(Wt + (size_t)(c4 * 4 + j) * 32 + 2 * r2) = d;
    }
  }
}

// ---------------- fused gated-FFN: out += diag(w) * (silu(X Wg) * (X Wu)) Wd ----------------
// One block = one 16-token tile; 8 waves, wave w owns output columns [w*16, w*16+16).
// Double-buffered LDS weight staging, software-pipelined: global loads for K-tile kt+1
// issue right after the barrier and retire underneath the WMMAs of tile kt.
template<int INTER, bool GATHER>
__global__ void ffn_kernel(const float* __restrict__ x,
                           const float* __restrict__ wg,
                           const float* __restrict__ wu,
                           const float* __restrict__ wd,
                           float* __restrict__ out,
                           const int* __restrict__ toks,
                           const float* __restrict__ tws,
                           const int* __restrict__ counts) {
  constexpr int NT1 = H_ / 32;     // 80
  constexpr int NT2 = INTER / 32;  // 52 or 104
  static_assert(NT1 % 2 == 0 && NT2 % 2 == 0, "even trip counts keep buffer parity safe");

  const int e    = blockIdx.y;
  const int tile = blockIdx.x;
  if (GATHER) { if (tile * 16 >= counts[e]) return; }

  const float* wgE = wg + (GATHER ? (size_t)e * H_ * INTER : (size_t)0);
  const float* wuE = wu + (GATHER ? (size_t)e * H_ * INTER : (size_t)0);
  const float* wdE = wd + (GATHER ? (size_t)e * INTER * H_ : (size_t)0);
  const int*   tokp = GATHER ? (toks + (size_t)e * T_ + tile * 16) : nullptr;
  const float* twp  = GATHER ? (tws  + (size_t)e * T_ + tile * 16) : nullptr;

  extern __shared__ unsigned short smem[];
  unsigned short* Xs  = smem;                  // 16 x H_     (bf16)
  unsigned short* Hsh = Xs + 16 * H_;          // 16 x INTER  (bf16)
  unsigned short* Wt0 = Hsh + 16 * INTER;      // 2 x (128 x 32), double buffer
  unsigned short* Wt1 = Wt0 + 2 * 128 * 32;    // 2 x (128 x 32), double buffer

  const int tid  = (int)threadIdx.x;
  const int lane = tid & 31;
  const int wave = tid >> 5;

  // ---- stage X tile (gathered, fp32 -> bf16, packed dword stores) ----
  for (int i = tid; i < 16 * (H_ / 4); i += 256) {
    const int row = i / (H_ / 4);
    const int c4  = i % (H_ / 4);
    const int tok = GATHER ? tokp[row] : (tile * 16 + row);
    const f32x4 v = ((const f32x4*)(x + (size_t)tok * H_))[c4];
    const u32x2 d = { pack2bf(v[0], v[1]), pack2bf(v[2], v[3]) };
    *(u32x2*)(Xs + row * H_ + c4 * 4) = d;
  }
  __syncthreads();

  // A-fragment addressing per the 16-bit 16x32 WMMA layout:
  // lanes 0-15 : row M=lane,    K = k+0..7  and k+16..23
  // lanes 16-31: row M=lane-16, K = k+8..15 and k+24..31
  const int mrow = lane & 15;
  const int kofs = (lane < 16) ? 0 : 8;
  const int rb   = (lane < 16) ? 0 : 8;    // B: K rows 0..7,16..23 vs 8..15,24..31
  const int bc   = wave * 16 + mrow;       // B column within 128-wide tile

  // ================= phase 1: Hsh = silu(X*Wg) * (X*Wu) =================
  for (int nb = 0; nb < INTER; nb += 128) {
    f32x8 accg = {}; f32x8 accuv = {};
    f32x4 pg[2][2], pu[2][2];
    load_pair<INTER>(wgE, 0, nb, tid, pg);
    load_pair<INTER>(wuE, 0, nb, tid, pu);
    for (int kt = 0; kt < NT1; ++kt) {
      unsigned short* wt0 = Wt0 + (kt & 1) * (128 * 32);
      unsigned short* wt1 = Wt1 + (kt & 1) * (128 * 32);
      store_tr(wt0, tid, pg);
      store_tr(wt1, tid, pu);
      __syncthreads();
      if (kt + 1 < NT1) {                       // prefetch next K-tile under the WMMAs
        load_pair<INTER>(wgE, (kt + 1) * 32, nb, tid, pg);
        load_pair<INTER>(wuE, (kt + 1) * 32, nb, tid, pu);
      }
      const int k = kt * 32;
      Frag16 a, bgf, buf;
      const unsigned short* ap = Xs + mrow * H_ + k + kofs;
      a.u[0] = *(const u32x4*)ap; a.u[1] = *(const u32x4*)(ap + 16);
      const unsigned short* bp0 = wt0 + bc * 32 + rb;
      const unsigned short* bp1 = wt1 + bc * 32 + rb;
      bgf.u[0] = *(const u32x4*)bp0; bgf.u[1] = *(const u32x4*)(bp0 + 16);
      buf.u[0] = *(const u32x4*)bp1; buf.u[1] = *(const u32x4*)(bp1 + 16);
      accg  = __builtin_amdgcn_wmma_f32_16x16x32_bf16(false, a.v, false, bgf.v, (short)0, accg,  false, false);
      accuv = __builtin_amdgcn_wmma_f32_16x16x32_bf16(false, a.v, false, buf.v, (short)0, accuv, false, false);
      // no trailing barrier: next iteration writes the other buffer; the single
      // barrier per iteration fences the buffer reused two iterations later.
    }
    // C/D layout: lane -> column (lane&15), VGPR v -> row v (+8 for lanes>=16)
    const int hc = nb + wave * 16 + mrow;
    const int mb = (lane < 16) ? 0 : 8;
#pragma unroll
    for (int vv = 0; vv < 8; ++vv)
      Hsh[(size_t)(mb + vv) * INTER + hc] = f2bf(silu_f(accg[vv]) * accuv[vv]);
  }

  // ================= phase 2: Out = Hsh * Wd, weighted scatter-add =================
  for (int nb = 0; nb < H_; nb += 128) {
    f32x8 acc = {};
    f32x4 pd[2][2];
    load_pair<H_>(wdE, 0, nb, tid, pd);
    for (int kt = 0; kt < NT2; ++kt) {
      unsigned short* wt = Wt0 + (kt & 1) * (128 * 32);
      store_tr(wt, tid, pd);
      __syncthreads();   // also fences phase-1 Hsh writes before first A-fragment read
      if (kt + 1 < NT2) load_pair<H_>(wdE, (kt + 1) * 32, nb, tid, pd);
      const int k = kt * 32;
      Frag16 a, b;
      const unsigned short* ap = Hsh + mrow * INTER + k + kofs;
      a.u[0] = *(const u32x4*)ap; a.u[1] = *(const u32x4*)(ap + 16);
      const unsigned short* bp = wt + bc * 32 + rb;
      b.u[0] = *(const u32x4*)bp; b.u[1] = *(const u32x4*)(bp + 16);
      acc = __builtin_amdgcn_wmma_f32_16x16x32_bf16(false, a.v, false, b.v, (short)0, acc, false, false);
    }
    const int col = nb + wave * 16 + mrow;
    const int mb  = (lane < 16) ? 0 : 8;
#pragma unroll
    for (int vv = 0; vv < 8; ++vv) {
      const int row = mb + vv;
      const int tok = GATHER ? tokp[row] : (tile * 16 + row);
      const float wgt = GATHER ? twp[row] : 1.0f;
      atomicAdd(out + (size_t)tok * H_ + col, wgt * acc[vv]);
    }
  }
}

// ---------------- launch ----------------
extern "C" void kernel_launch(void* const* d_in, const int* in_sizes, int n_in,
                              void* d_out, int out_size, void* d_ws, size_t ws_size,
                              hipStream_t stream) {
  (void)in_sizes; (void)n_in; (void)ws_size;
  const float* x  = (const float*)d_in[0];  // (B,S,H)
  const float* rw = (const float*)d_in[1];  // (E,H)
  const float* w1 = (const float*)d_in[2];  // (E,H,I)
  const float* w3 = (const float*)d_in[3];  // (E,H,I)
  const float* w2 = (const float*)d_in[4];  // (E,I,H)
  const float* sg = (const float*)d_in[5];  // (H,IS)
  const float* su = (const float*)d_in[6];  // (H,IS)
  const float* sd = (const float*)d_in[7];  // (IS,H)
  float* out = (float*)d_out;

  char* ws = (char*)d_ws;
  size_t off = 0;
  float* wcomb  = (float*)(ws + off); off += (size_t)T_ * E_ * sizeof(float);
  int*   counts = (int*)(ws + off);   off += 256;
  int*   toks   = (int*)(ws + off);   off += (size_t)E_ * T_ * sizeof(int);
  float* tws    = (float*)(ws + off); off += (size_t)E_ * T_ * sizeof(float);

  (void)hipMemsetAsync(d_out, 0, (size_t)out_size * sizeof(float), stream);

  router_kernel<<<T_ / 8, 256, 0, stream>>>(x, rw, wcomb);
  build_lists_kernel<<<E_, 32, 0, stream>>>(wcomb, counts, toks, tws);

  const size_t smemE = (size_t)(16 * H_ + 16 * I_  + 4 * 128 * 32) * sizeof(unsigned short);
  const size_t smemS = (size_t)(16 * H_ + 16 * IS_ + 4 * 128 * 32) * sizeof(unsigned short);
  (void)hipFuncSetAttribute(reinterpret_cast<const void*>(&ffn_kernel<I_,  true >),
                            hipFuncAttributeMaxDynamicSharedMemorySize, (int)smemE);
  (void)hipFuncSetAttribute(reinterpret_cast<const void*>(&ffn_kernel<IS_, false>),
                            hipFuncAttributeMaxDynamicSharedMemorySize, (int)smemS);

  // routed experts: grid (maxTiles=128, E); inactive tiles exit on counts[]
  ffn_kernel<I_,  true ><<<dim3(T_ / 16, E_), 256, smemE, stream>>>(x, w1, w3, w2, out, toks, tws, counts);
  // shared MLP: dense over all token tiles
  ffn_kernel<IS_, false><<<dim3(T_ / 16, 1 ), 256, smemS, stream>>>(x, sg, su, sd, out, nullptr, nullptr, nullptr);
}
